// NerfRenderer_66511863546378
// MI455X (gfx1250) — compile-verified, matrix operand order checked by reading, not run on hardware
//
#include <hip/hip_runtime.h>
#include <hip/hip_bf16.h>
#include <math.h>

typedef __attribute__((ext_vector_type(16))) _Float16 v16h;
typedef __attribute__((ext_vector_type(8)))  _Float16 v8h;
typedef __attribute__((ext_vector_type(8)))  float    v8f;

#define NSAMP 200
#define GRIDN 128

__device__ __forceinline__ v16h combine16(v8h lo, v8h hi) {
  return __builtin_shufflevector(lo, hi, 0,1,2,3,4,5,6,7,8,9,10,11,12,13,14,15);
}

__device__ __forceinline__ v8f wmma16(v16h a, v16h b, v8f c) {
  return __builtin_amdgcn_wmma_f32_16x16x32_f16(false, a, false, b, (short)0, c, false, false);
}

__device__ __forceinline__ v8f bcast8(float x) {
  v8f c;
#pragma unroll
  for (int r = 0; r < 8; ++r) c[r] = x;
  return c;
}

__device__ __forceinline__ unsigned ballot32(bool p) {
#if __has_builtin(__builtin_amdgcn_ballot_w32)
  return __builtin_amdgcn_ballot_w32(p);
#else
  return (unsigned)__ballot(p);
#endif
}

// swap values between lane L and lane L+16 (SWAPX16 ds_swizzle pattern)
__device__ __forceinline__ float swap_half(float x) {
  int i = __builtin_amdgcn_ds_swizzle(__float_as_int(x), 0x401F);
  return __int_as_float(i);
}

// Build one 32x16 f16 B tile (WMMA SRC1 layout) in registers from a row-major
// f32 weight matrix W[rows][cols]. Branchless: clamped load + select-zero.
__device__ __forceinline__ v16h load_b_tile_global(const float* __restrict__ W, int rows,
                                                   int cols, int kBase, int nBase, int lane) {
  const int half = lane >> 4;
  const int n = nBase + (lane & 15);
  const int nc = min(n, cols - 1);
  v16h out;
#pragma unroll
  for (int j = 0; j < 16; ++j) {
    int koff = (j < 8) ? (half * 8 + j) : (16 + half * 8 + (j - 8));
    int k = kBase + koff;
    int kc = min(k, rows - 1);
    float v = W[kc * cols + nc];
    v = (k < rows && n < cols) ? v : 0.0f;
    out[j] = (_Float16)v;
  }
  return out;
}

// Load a 16x32 f16 A tile (WMMA SRC0 layout) from an LDS-staged row-major matrix.
__device__ __forceinline__ v16h load_a_tile(const _Float16* stg, int rowStride,
                                            int colBase, int lane) {
  const int row = lane & 15;
  const int c0 = colBase + ((lane >> 4) << 3);
  const _Float16* p = stg + row * rowStride + c0;
  v8h lo = *(const v8h*)(p);
  v8h hi = *(const v8h*)(p + 16);
  return combine16(lo, hi);
}

// Store an f32 C tile (N across lanes, M across vgpr+lanehalf) into LDS as f16 row-major.
__device__ __forceinline__ void store_c_f16(_Float16* stg, int rowStride, int nBase,
                                            int lane, v8f c) {
  const int col = nBase + (lane & 15);
  const int r0 = (lane >> 4) << 3;
#pragma unroll
  for (int r = 0; r < 8; ++r)
    stg[(r0 + r) * rowStride + col] = (_Float16)c[r];
}

// Load a pre-linearized B tile from LDS (tile stored as [lane][16] halves).
__device__ __forceinline__ v16h load_ldsb(const _Float16* ldsB, int tile, int lane) {
  const _Float16* p = ldsB + tile * 512 + lane * 16;
  v8h lo = *(const v8h*)p;
  v8h hi = *(const v8h*)(p + 8);
  return combine16(lo, hi);
}

struct __align__(16) WaveScratch {
  _Float16 stageIn[16 * 32];   // K<=32-padded inputs (dirs once, then sc each step)
  _Float16 stageH[16 * 64];    // h / feat / rh staging
  float    sig[16];
  float    rgbx[16][4];
};

#define DSWAIT() asm volatile("s_wait_dscnt 0x0" ::: "memory")

__global__ __launch_bounds__(256)
void nerf_render_kernel(const float* __restrict__ rays_o, const float* __restrict__ rays_d,
                        const float* __restrict__ grid,   const float* __restrict__ aabb,
                        const float* __restrict__ fw1,    const float* __restrict__ fb1,
                        const float* __restrict__ fw2,    const float* __restrict__ fb2,
                        const float* __restrict__ sw,     const float* __restrict__ sb,
                        const float* __restrict__ rw1,    const float* __restrict__ rb1,
                        const float* __restrict__ rw2,    const float* __restrict__ rb2,
                        float* __restrict__ out, int n_rays)
{
  __shared__ _Float16 ldsB[8 * 512];     // 8 small B tiles: fw1 x4, sw x2, rw2 x2
  __shared__ WaveScratch wsmem[8];

  const int tid  = threadIdx.x;
  const int lane = tid & 31;
  const int wave = tid >> 5;
  const int ncol = lane & 15;
  const int half = lane >> 4;

  // ---- block-cooperative: small weights -> LDS in B-tile-linear layout ----
  {
    const int tile = tid >> 5;
    const float* W; int rows, cols, kb, nb;
    if (tile < 4)      { W = fw1; rows = 3;  cols = 64; kb = 0;               nb = 16 * tile; }
    else if (tile < 6) { W = sw;  rows = 64; cols = 1;  kb = 32 * (tile - 4); nb = 0; }
    else               { W = rw2; rows = 64; cols = 3;  kb = 32 * (tile - 6); nb = 0; }
    const int n = nb + ncol;
    const int nc = min(n, cols - 1);
    _Float16* dst = &ldsB[tile * 512 + lane * 16];
#pragma unroll
    for (int j = 0; j < 16; ++j) {
      int koff = (j < 8) ? (half * 8 + j) : (16 + half * 8 + (j - 8));
      int k = kb + koff;
      int kc = min(k, rows - 1);
      float v = W[kc * cols + nc];
      v = (k < rows && n < cols) ? v : 0.0f;
      dst[j] = (_Float16)v;
    }
  }
  __syncthreads();

  WaveScratch& S = wsmem[wave];
  const int rayBase = (blockIdx.x * 8 + wave) * 16;

  // ---- hot weights to registers (per wave): fw2 and rw1 feat-chunks ----
  v16h Bw2[4][2], Brw1[4][2];
#pragma unroll
  for (int t = 0; t < 4; ++t)
#pragma unroll
    for (int c = 0; c < 2; ++c) {
      Bw2[t][c]  = load_b_tile_global(fw2, 64, 64, 32 * c, 16 * t, lane);
      Brw1[t][c] = load_b_tile_global(rw1, 67, 64, 32 * c, 16 * t, lane);
    }

  // per-lane biases (C layout: lane holds column n = 16*t + ncol)
  float b1v[4], b2v[4], rb1v[4];
#pragma unroll
  for (int t = 0; t < 4; ++t) {
    b1v[t]  = fb1[16 * t + ncol];
    b2v[t]  = fb2[16 * t + ncol];
    rb1v[t] = rb1[16 * t + ncol];
  }
  const float sbv  = sb[0];
  const float rb2v = (ncol < 3) ? rb2[ncol] : 0.0f;

  // ---- ray setup (lane L and L+16 both own ray rayBase+ncol) ----
  const float a0x = aabb[0], a0y = aabb[1], a0z = aabb[2];
  const float a1x = aabb[3], a1y = aabb[4], a1z = aabb[5];
  const float ex = a1x - a0x, ey = a1y - a0y, ez = a1z - a0z;
  const float step = sqrtf(ex * ex + ey * ey + ez * ez) / (float)NSAMP;
  const float sclx = 2.0f / ex, scly = 2.0f / ey, sclz = 2.0f / ez;

  int ray = rayBase + ncol;
  if (ray >= n_rays) ray = n_rays - 1;
  const float rox = rays_o[ray * 3 + 0], roy = rays_o[ray * 3 + 1], roz = rays_o[ray * 3 + 2];
  const float rdx = rays_d[ray * 3 + 0], rdy = rays_d[ray * 3 + 1], rdz = rays_d[ray * 3 + 2];

  float tmin;
  {
    float ddx = (rdx == 0.0f) ? 1e-9f : rdx;
    float ddy = (rdy == 0.0f) ? 1e-9f : rdy;
    float ddz = (rdz == 0.0f) ? 1e-9f : rdz;
    float mx = fminf((a0x - rox) / ddx, (a1x - rox) / ddx);
    float my = fminf((a0y - roy) / ddy, (a1y - roy) / ddy);
    float mz = fminf((a0z - roz) / ddz, (a1z - roz) / ddz);
    tmin = fmaxf(fmaxf(mx, my), mz);
    tmin = fminf(fmaxf(tmin, 0.0f), 100000.0f);
  }

  // ---- stage dirs once, build A_dir, precompute rgb-hidden direction bias ----
  {
    _Float16* p = &S.stageIn[lane * 16];
#pragma unroll
    for (int j = 0; j < 16; ++j) p[j] = (_Float16)0.0f;   // zero whole 16x32 region
  }
  DSWAIT();
  if (lane < 16) {
    S.stageIn[lane * 32 + 0] = (_Float16)rdx;
    S.stageIn[lane * 32 + 1] = (_Float16)rdy;
    S.stageIn[lane * 32 + 2] = (_Float16)rdz;
  }
  DSWAIT();
  const v16h Adir = load_a_tile(S.stageIn, 32, 0, lane);

  v8f rcb[4];   // rcb[t] = dirs @ rw1[64:67] + rb1 (C layout, loop-invariant)
#pragma unroll
  for (int t = 0; t < 4; ++t) {
    v16h Bd = load_b_tile_global(rw1, 67, 64, 64, 16 * t, lane);
    rcb[t] = wmma16(Adir, Bd, bcast8(rb1v[t]));
  }

  // ---- march ----
  float trans = 1.0f, accR = 0.0f, accG = 0.0f, accB = 0.0f;
  const int r0lane = half << 3;

  for (int s = 0; s < NSAMP; ++s) {
    // all rays opaque -> no further contribution possible (trans monotone)
    const unsigned alive = ballot32((lane < 16) && (trans > 1e-4f));
    if (alive == 0u) break;

    // sample point + occupancy; both half-waves compute, corners split by z
    float t  = tmin + (float)s * step;
    float px = rox + rdx * t, py = roy + rdy * t, pz = roz + rdz * t;
    bool inb = (px >= a0x) & (px <= a1x) & (py >= a0y) & (py <= a1y) &
               (pz >= a0z) & (pz <= a1z);
    float scx = (px - a0x) * sclx - 1.0f;
    float scy = (py - a0y) * scly - 1.0f;
    float scz = (pz - a0z) * sclz - 1.0f;
    // trilinear occupancy (torch grid_sample, align_corners=False, zero pad)
    float gxf = ((scx + 1.0f) * (float)GRIDN - 1.0f) * 0.5f;
    float gyf = ((scy + 1.0f) * (float)GRIDN - 1.0f) * 0.5f;
    float gzf = ((scz + 1.0f) * (float)GRIDN - 1.0f) * 0.5f;
    float x0 = floorf(gxf), y0 = floorf(gyf), z0 = floorf(gzf);
    float tri;
    {
      // this half-wave handles z-corner cz = half
      float zi = z0 + (float)half;
      float wz = 1.0f - fabsf(gzf - zi);
      bool  vz = (zi >= 0.0f) & (zi < (float)GRIDN);
      int   zc = min(max((int)zi, 0), GRIDN - 1);
      float acc = 0.0f;
#pragma unroll
      for (int cy = 0; cy < 2; ++cy)
#pragma unroll
        for (int cx = 0; cx < 2; ++cx) {
          float xi = x0 + (float)cx, yi = y0 + (float)cy;
          float w = (1.0f - fabsf(gxf - xi)) * (1.0f - fabsf(gyf - yi)) * wz;
          bool valid = vz & (xi >= 0.0f) & (xi < (float)GRIDN) &
                       (yi >= 0.0f) & (yi < (float)GRIDN);
          int xc = min(max((int)xi, 0), GRIDN - 1);
          int yc = min(max((int)yi, 0), GRIDN - 1);
          float gv = grid[(zc * GRIDN + yc) * GRIDN + xc];
          acc += valid ? (w * gv) : 0.0f;
        }
      tri = acc + swap_half(acc);   // combine the two z-halves
    }
    const bool m = inb && (tri > 0.01f);
    const unsigned mb = ballot32((lane < 16) && m) & 0xFFFFu;
    // if no *alive* ray hits occupied space, this step is a provable no-op
    if ((mb & alive) == 0u) continue;

    // stage sc (cols 3..31 stay zero from init)
    if (lane < 16) {
      S.stageIn[lane * 32 + 0] = (_Float16)scx;
      S.stageIn[lane * 32 + 1] = (_Float16)scy;
      S.stageIn[lane * 32 + 2] = (_Float16)scz;
    }
    DSWAIT();
    const v16h Asc = load_a_tile(S.stageIn, 32, 0, lane);

    // layer 1: h = relu(sc @ fw1 + fb1)
#pragma unroll
    for (int tt = 0; tt < 4; ++tt) {
      v8f c = wmma16(Asc, load_ldsb(ldsB, tt, lane), bcast8(b1v[tt]));
#pragma unroll
      for (int r = 0; r < 8; ++r) c[r] = fmaxf(c[r], 0.0f);
      store_c_f16(S.stageH, 64, 16 * tt, lane, c);
    }
    DSWAIT();
    const v16h Ah0 = load_a_tile(S.stageH, 64, 0, lane);
    const v16h Ah1 = load_a_tile(S.stageH, 64, 32, lane);

    // layer 2: feat = mask ? (h @ fw2 + fb2) : 0   (select, NaN-safe)
#pragma unroll
    for (int tt = 0; tt < 4; ++tt) {
      v8f c = bcast8(b2v[tt]);
      c = wmma16(Ah0, Bw2[tt][0], c);
      c = wmma16(Ah1, Bw2[tt][1], c);
#pragma unroll
      for (int r = 0; r < 8; ++r) c[r] = ((mb >> (r0lane + r)) & 1u) ? c[r] : 0.0f;
      store_c_f16(S.stageH, 64, 16 * tt, lane, c);
    }
    DSWAIT();
    const v16h Af0 = load_a_tile(S.stageH, 64, 0, lane);
    const v16h Af1 = load_a_tile(S.stageH, 64, 32, lane);

    // sigma = feat @ sw + sb  (N padded 1->16, only column 0 used)
    {
      v8f c = bcast8(sbv);
      c = wmma16(Af0, load_ldsb(ldsB, 4, lane), c);
      c = wmma16(Af1, load_ldsb(ldsB, 5, lane), c);
      if (ncol == 0) {
#pragma unroll
        for (int r = 0; r < 8; ++r) S.sig[r0lane + r] = c[r];
      }
    }

    // rgb hidden: rh = relu(feat @ rw1[:64] + (dirs @ rw1[64:] + rb1))
#pragma unroll
    for (int tt = 0; tt < 4; ++tt) {
      v8f c = rcb[tt];
      c = wmma16(Af0, Brw1[tt][0], c);
      c = wmma16(Af1, Brw1[tt][1], c);
#pragma unroll
      for (int r = 0; r < 8; ++r) c[r] = fmaxf(c[r], 0.0f);
      store_c_f16(S.stageH, 64, 16 * tt, lane, c);
    }
    DSWAIT();
    const v16h Ar0 = load_a_tile(S.stageH, 64, 0, lane);
    const v16h Ar1 = load_a_tile(S.stageH, 64, 32, lane);

    // rgb out: rgb_logits = rh @ rw2 + rb2 (N padded 3->16)
    {
      v8f c = bcast8(rb2v);
      c = wmma16(Ar0, load_ldsb(ldsB, 6, lane), c);
      c = wmma16(Ar1, load_ldsb(ldsB, 7, lane), c);
      if (ncol < 3) {
#pragma unroll
        for (int r = 0; r < 8; ++r) S.rgbx[r0lane + r][ncol] = c[r];
      }
    }
    DSWAIT();

    // alpha compositing (one ray per lane 0..15)
    if (lane < 16) {
      float sg    = m ? S.sig[lane] : 0.0f;
      float ea    = __expf(-sg * step);
      float alpha = 1.0f - ea;
      float w     = trans * alpha;
      if (m && (trans > 1e-4f)) {
        float r_ = 1.0f / (1.0f + __expf(-S.rgbx[lane][0]));
        float g_ = 1.0f / (1.0f + __expf(-S.rgbx[lane][1]));
        float b_ = 1.0f / (1.0f + __expf(-S.rgbx[lane][2]));
        accR += r_ * w; accG += g_ * w; accB += b_ * w;
      }
      trans *= ea;
    }
  }

  if (lane < 16 && (rayBase + lane) < n_rays) {
    const int r = rayBase + lane;
    out[r * 3 + 0] = accR;
    out[r * 3 + 1] = accG;
    out[r * 3 + 2] = accB;
  }
}

extern "C" void kernel_launch(void* const* d_in, const int* in_sizes, int n_in,
                              void* d_out, int out_size, void* d_ws, size_t ws_size,
                              hipStream_t stream) {
  const float* rays_o = (const float*)d_in[0];
  const float* rays_d = (const float*)d_in[1];
  const float* grid   = (const float*)d_in[2];
  const float* aabb   = (const float*)d_in[3];
  const float* fw1    = (const float*)d_in[4];
  const float* fb1    = (const float*)d_in[5];
  const float* fw2    = (const float*)d_in[6];
  const float* fb2    = (const float*)d_in[7];
  const float* sw     = (const float*)d_in[8];
  const float* sb     = (const float*)d_in[9];
  const float* rw1    = (const float*)d_in[10];
  const float* rb1    = (const float*)d_in[11];
  const float* rw2    = (const float*)d_in[12];
  const float* rb2    = (const float*)d_in[13];
  float* out = (float*)d_out;

  const int n_rays = in_sizes[0] / 3;           // 8192
  const int rays_per_block = 8 * 16;            // 8 waves x 16 rays
  const int blocks = (n_rays + rays_per_block - 1) / rays_per_block;

  nerf_render_kernel<<<blocks, 256, 0, stream>>>(
      rays_o, rays_d, grid, aabb, fw1, fb1, fw2, fb2, sw, sb,
      rw1, rb1, rw2, rb2, out, n_rays);
}
